// JTNNDecoder_14946486190823
// MI455X (gfx1250) — matrix operand order repaired and checked by reading, not compiled, not run on hardware
//
#include <hip/hip_runtime.h>

#define HIDS   512
#define LATS   256
#define FTOT   768            // HID + LAT
#define VOCAB  780
#define VTILES 52             // vocab padded to 52 tiles of 16
#define VPAD   (VTILES * 16)  // 832 rows in padded Wo
#define MT     64             // rows per workgroup (4 M-blocks of 16)
#define KC     384            // K-chunk staged in LDS (2 chunks cover 768)
#define ASTRIDE (KC + 8)      // 392 elems = 784B/row (bank offset 4, 16B-aligned)
#define HSTRIDE (HIDS + 8)    // 520 elems = 1040B/row (bank offset 4, 16B-aligned)

typedef __attribute__((ext_vector_type(16))) __bf16 v16bf;
typedef __attribute__((ext_vector_type(8)))  __bf16 v8bf;
typedef __attribute__((ext_vector_type(8)))  float  v8f;

union Frag { v16bf v; v8bf h[2]; };

static __device__ __forceinline__ unsigned short f2bf(float f) {
    unsigned u = __float_as_uint(f);
    unsigned r = ((u >> 16) & 1u) + 0x7FFFu;   // round-to-nearest-even
    return (unsigned short)((u + r) >> 16);
}

// ---- prologue: convert W_w [512][768] f32 -> bf16 (same layout) ----
__global__ void convert_Ww(const float* __restrict__ w, unsigned short* __restrict__ o, int n) {
    int i = blockIdx.x * 256 + threadIdx.x;
    if (i < n) o[i] = f2bf(w[i]);
}

// ---- prologue: convert Wo_w [780][512] f32 -> bf16 zero-padded to [832][512] ----
__global__ void convert_Wo(const float* __restrict__ w, unsigned short* __restrict__ o) {
    int i = blockIdx.x * 256 + threadIdx.x;           // grid covers VPAD*HIDS
    int v = i >> 9, k = i & 511;
    float val = (v < VOCAB) ? w[v * HIDS + k] : 0.0f;
    o[i] = f2bf(val);
}

// ---- fused: gather + concat -> GEMM1(relu,bias) -> GEMM2(bias) ----
__global__ void __launch_bounds__(256)
decoder_fused(const float* __restrict__ hiddens,
              const int*   __restrict__ contexts,
              const float* __restrict__ xtree,
              const unsigned short* __restrict__ WwB,   // [512][768] bf16
              const float* __restrict__ Wb,             // [512] f32
              const unsigned short* __restrict__ WoB,   // [832][512] bf16 (rows>=780 zero)
              const float* __restrict__ Wob,            // [780] f32
              float* __restrict__ out,                  // [N][780] f32
              int nrows)
{
    __shared__ unsigned short Atile[MT * ASTRIDE];   // 50,176 B
    __shared__ unsigned short Htile[MT * HSTRIDE];   // 66,560 B  (116.7 KB -> 2 WG/WGP)

    const int t    = threadIdx.x;
    const int row0 = blockIdx.x * MT;

    const int lane = t & 31;
    const int w    = t >> 5;         // wave id 0..7
    const int hi   = lane >> 4;      // K-half select (this code's contiguous k = 16*hi + s order)
    const int ln   = lane & 15;      // M (A) / N (B,C,D) index within tile

    // ================= GEMM1: H = relu(concat(A) x Ww^T + b) =================
    // Each wave: 4 N-tiles (cols [64w,64w+64)) x ALL 4 M-blocks.
    // => every B fragment feeds 4 WMMAs; W_w is read exactly once per WG.
    v8f acc[4][4] = {};   // [nt][mb], 128 VGPRs

    // single base pointers; all displacements are compile-time constants
    const unsigned short* bbase = WwB + (size_t)(64 * w + ln) * FTOT + hi * 16;
    const unsigned short* albase = &Atile[ln * ASTRIDE + hi * 16];

    for (int kc = 0; kc < 2; ++kc) {
        __syncthreads();             // previous chunk fully consumed
        // ---- stage 64 x 384 bf16 chunk of concat(hiddens, xtree[ctx]) ----
        #pragma unroll 4
        for (int i = 0; i < 24; ++i) {                   // 6144 float4 / 256 thr
            int idx = t + i * 256;
            int r = idx / 96, c4 = idx % 96;
            int gr = row0 + r; if (gr >= nrows) gr = nrows - 1;
            int cg = kc * KC + (c4 << 2);                // global feature col
            float4 f;
            if (cg < HIDS) {
                f = reinterpret_cast<const float4*>(hiddens + (size_t)gr * HIDS)[cg >> 2];
            } else {
                int ctx = contexts[gr];
                f = reinterpret_cast<const float4*>(xtree + (size_t)ctx * LATS)[(cg - HIDS) >> 2];
            }
            *reinterpret_cast<ushort4*>(&Atile[r * ASTRIDE + (c4 << 2)]) =
                make_ushort4(f2bf(f.x), f2bf(f.y), f2bf(f.z), f2bf(f.w));
        }
        __syncthreads();

        // ---- 12 k-steps: 4 A frags + 4 B frags -> 16 WMMAs ----
        for (int kk = 0; kk < KC; kk += 32) {
            const unsigned short* ak = albase + kk;
            const unsigned short* bk = bbase + kc * KC + kk;
            Frag a[4];
            #pragma unroll
            for (int mb = 0; mb < 4; ++mb) {             // mb*16 rows: +mb*12544B (imm)
                a[mb].h[0] = *reinterpret_cast<const v8bf*>(ak + mb * 16 * ASTRIDE);
                a[mb].h[1] = *reinterpret_cast<const v8bf*>(ak + mb * 16 * ASTRIDE + 8);
            }
            #pragma unroll
            for (int nt = 0; nt < 4; ++nt) {
                Frag b;                                  // +nt*24576B (imm offset)
                b.v = *reinterpret_cast<const v16bf*>(bk + (size_t)nt * 16 * FTOT);
                #pragma unroll
                for (int mb = 0; mb < 4; ++mb)
                    acc[nt][mb] = __builtin_amdgcn_wmma_f32_16x16x32_bf16(
                        false, a[mb].v, false, b.v, (short)0, acc[nt][mb], false, false);
            }
        }
    }

    // ---- bias + ReLU -> Htile (bf16) ----
    #pragma unroll
    for (int nt = 0; nt < 4; ++nt) {
        int c = 64 * w + nt * 16 + ln;
        float bias = Wb[c];
        #pragma unroll
        for (int mb = 0; mb < 4; ++mb) {
            #pragma unroll
            for (int j = 0; j < 8; ++j) {
                int r = mb * 16 + j + hi * 8;            // C/D VGPR layout
                float v = acc[nt][mb][j] + bias;
                Htile[r * HSTRIDE + c] = f2bf(v > 0.0f ? v : 0.0f);
            }
        }
    }
    __syncthreads();

    // ================= GEMM2: out = H x Wo^T + b =================
    // Each wave: a pair of vocab tiles x all 4 M-blocks per pass (B reuse x4).
    const unsigned short* hbase = &Htile[ln * HSTRIDE + hi * 16];
    const unsigned short* obase = WoB + (size_t)ln * HIDS + hi * 16;

    for (int p = w; p < (VTILES / 2); p += 8) {          // pair p -> tiles 2p, 2p+1
        v8f oacc[2][4] = {};                             // [q][mb], 64 VGPRs
        const unsigned short* opb = obase + (size_t)p * 32 * HIDS;
        for (int k0 = 0; k0 < HIDS; k0 += 32) {
            Frag a[4];
            #pragma unroll
            for (int mb = 0; mb < 4; ++mb) {             // +mb*16640B (imm DS offset)
                a[mb].h[0] = *reinterpret_cast<const v8bf*>(hbase + k0 + mb * 16 * HSTRIDE);
                a[mb].h[1] = *reinterpret_cast<const v8bf*>(hbase + k0 + mb * 16 * HSTRIDE + 8);
            }
            #pragma unroll
            for (int q = 0; q < 2; ++q) {
                Frag b;                                  // +q*16384B (imm offset)
                b.v = *reinterpret_cast<const v16bf*>(opb + (size_t)q * 16 * HIDS + k0);
                #pragma unroll
                for (int mb = 0; mb < 4; ++mb)
                    oacc[q][mb] = __builtin_amdgcn_wmma_f32_16x16x32_bf16(
                        false, a[mb].v, false, b.v, (short)0, oacc[q][mb], false, false);
            }
        }
        #pragma unroll
        for (int q = 0; q < 2; ++q) {
            int c = (2 * p + q) * 16 + ln;
            if (c < VOCAB) {
                float bias = Wob[c];
                #pragma unroll
                for (int mb = 0; mb < 4; ++mb) {
                    #pragma unroll
                    for (int j = 0; j < 8; ++j) {
                        int gr = row0 + mb * 16 + j + hi * 8;
                        if (gr < nrows)
                            out[(size_t)gr * VOCAB + c] = oacc[q][mb][j] + bias;
                    }
                }
            }
        }
    }
}

extern "C" void kernel_launch(void* const* d_in, const int* in_sizes, int n_in,
                              void* d_out, int out_size, void* d_ws, size_t ws_size,
                              hipStream_t stream) {
    const float* hiddens  = (const float*)d_in[0];
    const int*   contexts = (const int*)  d_in[1];   // jax int64 w/o x64 -> int32
    const float* xtree    = (const float*)d_in[2];
    const float* Ww       = (const float*)d_in[3];
    const float* Wb       = (const float*)d_in[4];
    const float* Wo       = (const float*)d_in[5];
    const float* Wob      = (const float*)d_in[6];
    float* out = (float*)d_out;
    const int nrows = in_sizes[0] / HIDS;

    unsigned short* WwB = (unsigned short*)d_ws;             // 512*768*2 = 786,432 B
    unsigned short* WoB = WwB + HIDS * FTOT;                 // 832*512*2 = 851,968 B

    convert_Ww<<<(HIDS * FTOT + 255) / 256, 256, 0, stream>>>(Ww, WwB, HIDS * FTOT);
    convert_Wo<<<(VPAD * HIDS + 255) / 256, 256, 0, stream>>>(Wo, WoB);

    const int ngrid = (nrows + MT - 1) / MT;
    decoder_fused<<<ngrid, 256, 0, stream>>>(hiddens, contexts, xtree,
                                             WwB, Wb, WoB, Wob, out, nrows);
}
